// BertUnpadSelfAttention_54820962566235
// MI455X (gfx1250) — matrix operand order repaired and checked by reading, not compiled, not run on hardware
//
#include <hip/hip_runtime.h>
#include <hip/hip_bf16.h>

// BertUnpadSelfAttention for MI455X (gfx1250, wave32, WMMA bf16).
// Pipeline: [memset QKV pads] -> prep_w (fp32 W -> bf16 W^T) -> prep_x (fp32 X -> bf16)
//           -> qkv_gemm (WMMA, scatter to padded Q,K [s][d] and V^T [d][s])
//           -> attn (flash-attention, WMMA QK^T and P.V, online softmax)

#define Bn   8
#define Sn   1024
#define HIDn 768
#define Hn   12
#define Dn   64
#define NNZ  6528
#define NCOL 2304   // 3*HID

typedef __attribute__((ext_vector_type(16))) __bf16 v16bf;
typedef __attribute__((ext_vector_type(8)))  __bf16 v8bf;
typedef __attribute__((ext_vector_type(4)))  __bf16 v4bf;
typedef __attribute__((ext_vector_type(8)))  float  v8f;

static __device__ inline v8f wmma_bf16(v16bf a, v16bf b, v8f c) {
    return __builtin_amdgcn_wmma_f32_16x16x32_bf16(
        false, a, false, b, (short)0, c, false, false);
}

static __device__ inline v16bf cat16(v8bf lo, v8bf hi) {
    return __builtin_shufflevector(lo, hi, 0,1,2,3,4,5,6,7,8,9,10,11,12,13,14,15);
}

// ---- prep: W[768][2304] fp32 -> Wt[2304][768] bf16 (LDS tile transpose) ----
__global__ __launch_bounds__(256)
void prep_w(const float* __restrict__ W, __bf16* __restrict__ Wt)
{
    __shared__ float tile[32][33];
    const int tx = threadIdx.x & 31, ty = threadIdx.x >> 5;   // 32 x 8
    const int c0 = blockIdx.x * 32;                            // over NCOL (72)
    const int r0 = blockIdx.y * 32;                            // over HIDn (24)
    #pragma unroll
    for (int j = 0; j < 4; ++j)
        tile[ty + j * 8][tx] = W[(r0 + ty + j * 8) * NCOL + c0 + tx];
    __syncthreads();
    #pragma unroll
    for (int j = 0; j < 4; ++j)
        Wt[(size_t)(c0 + ty + j * 8) * HIDn + r0 + tx] = (__bf16)tile[tx][ty + j * 8];
}

// ---- prep: X fp32 -> bf16, 4 elements per thread ----
__global__ __launch_bounds__(256)
void prep_x(const float* __restrict__ X, __bf16* __restrict__ Xb)
{
    const int i = blockIdx.x * blockDim.x + threadIdx.x;       // 1,253,376 quads
    const float4 v = ((const float4*)X)[i];
    v4bf o; o[0] = (__bf16)v.x; o[1] = (__bf16)v.y; o[2] = (__bf16)v.z; o[3] = (__bf16)v.w;
    ((v4bf*)Xb)[i] = o;
}

// ---- Kernel 1: QKV = Xb @ Wt^T + b ; scatter to Q,K[s][d], Vt[d][s] (bf16) ----
__global__ __launch_bounds__(256)
void qkv_gemm(const __bf16* __restrict__ Xb, const __bf16* __restrict__ Wt,
              const float* __restrict__ bW, const int* __restrict__ idx,
              __bf16* __restrict__ Qb, __bf16* __restrict__ Kb, __bf16* __restrict__ Vt)
{
    const int wave = (blockIdx.x * blockDim.x + threadIdx.x) >> 5;
    const int COLT = NCOL / 64;                 // 36 col tiles of 64
    if (wave >= (NNZ / 16) * COLT) return;
    const int rowTile = wave / COLT;
    const int colTile = wave % COLT;

    const int lane = threadIdx.x & 31;
    const int n16  = lane & 15;
    const int aKh  = (lane & 16) ? 16 : 0;
    const int bKh  = (lane & 16) ? 16 : 0;
    const int mOff = (lane & 16) ? 8  : 0;

    v8f acc[4] = {};
    const __bf16* arow = Xb + (size_t)(rowTile * 16 + n16) * HIDn;

    for (int k0 = 0; k0 < HIDn; k0 += 32) {
        const __bf16* ap = arow + k0 + aKh;
        const v16bf a = cat16(*(const v8bf*)ap, *(const v8bf*)(ap + 8));
        #pragma unroll
        for (int c4 = 0; c4 < 4; ++c4) {
            const int n = colTile * 64 + c4 * 16 + n16;
            const v16bf bb = *(const v16bf*)(Wt + (size_t)n * HIDn + k0 + bKh);
            acc[c4] = wmma_bf16(a, bb, acc[c4]);
        }
    }

    #pragma unroll
    for (int r = 0; r < 8; ++r) {
        const int flat = idx[rowTile * 16 + r + mOff];
        const int bb = flat >> 10;
        const int ss = flat & 1023;
        #pragma unroll
        for (int c4 = 0; c4 < 4; ++c4) {
            const int col = colTile * 64 + c4 * 16 + n16;
            const float v = acc[c4][r] + bW[col];
            const int t   = col / HIDn;
            const int rem = col % HIDn;
            const int hh  = rem >> 6;
            const int dd  = rem & 63;
            if (t == 2)   // V stored transposed: [b][h][d][s]
                Vt[(((size_t)(bb * Hn + hh) * Dn) + dd) * Sn + ss] = (__bf16)v;
            else {
                __bf16* dst = (t == 0) ? Qb : Kb;
                dst[(((size_t)(bb * Hn + hh) * Sn) + ss) * Dn + dd] = (__bf16)v;
            }
        }
    }
}

// ---- Kernel 2: flash attention, one wave per (b, h, 16-row q tile) ----
__global__ __launch_bounds__(256)
void attn(const __bf16* __restrict__ Qb, const __bf16* __restrict__ Kb,
          const __bf16* __restrict__ Vt, const float* __restrict__ bias,
          const int* __restrict__ cu, float* __restrict__ out)
{
    __shared__ __bf16 pstage[8][16 * 32];
    const int wib = threadIdx.x >> 5;
    const int w   = blockIdx.x * 8 + wib;
    const int QT  = Sn / 16;
    const int b   = w / (Hn * QT);
    const int rem = w % (Hn * QT);
    const int h   = rem / QT;
    const int q0  = (rem % QT) * 16;

    const int base = cu[b];
    const int len  = cu[b + 1] - base;
    if (q0 >= len) return;

    const int lane = threadIdx.x & 31;
    const int n16  = lane & 15;
    const int aKh  = (lane & 16) ? 16 : 0;
    const int bKh  = (lane & 16) ? 16 : 0;
    const int mOff = (lane & 16) ? 8  : 0;

    const __bf16* Qp = Qb + (size_t)((b * Hn + h) * Sn) * Dn;
    const __bf16* Kp = Kb + (size_t)((b * Hn + h) * Sn) * Dn;
    const __bf16* Vp = Vt + (size_t)((b * Hn + h) * Dn) * Sn;
    const float* biasP = bias + ((size_t)(b * Hn + h) * Sn + q0) * Sn;

    // Q A-fragments (d 0..31 and 32..63), contiguous 16B chunks
    const __bf16* qp = Qp + (size_t)(q0 + n16) * Dn + aKh;
    const v16bf qa0 = cat16(*(const v8bf*)qp,        *(const v8bf*)(qp + 8));
    const v16bf qa1 = cat16(*(const v8bf*)(qp + 32), *(const v8bf*)(qp + 40));

    v8f acc[4] = {};
    float mrow[8], lrow[8];
    #pragma unroll
    for (int r = 0; r < 8; ++r) { mrow[r] = -1e30f; lrow[r] = 0.f; }

    for (int kt = 0; kt < Sn / 32; ++kt) {
        const int k0 = kt * 32;

        // scores: two 16x16 tiles over 32 keys, K-dim = D = 64
        v8f s0 = {}, s1 = {};
        {
            const __bf16* kp0 = Kp + (size_t)(k0 + n16) * Dn + bKh;
            s0 = wmma_bf16(qa0, *(const v16bf*)kp0,        s0);
            s0 = wmma_bf16(qa1, *(const v16bf*)(kp0 + 32), s0);
            const __bf16* kp1 = kp0 + 16 * Dn;
            s1 = wmma_bf16(qa0, *(const v16bf*)kp1,        s1);
            s1 = wmma_bf16(qa1, *(const v16bf*)(kp1 + 32), s1);
        }

        // online softmax (rows live in 16-lane halves; xor-reduce stays in-half)
        float csc[8], p0[8], p1[8];
        #pragma unroll
        for (int r = 0; r < 8; ++r) {
            const int qm = r + mOff;
            const float v0 = s0[r] * 0.125f + biasP[qm * Sn + k0 + n16];
            const float v1 = s1[r] * 0.125f + biasP[qm * Sn + k0 + 16 + n16];
            float mt = fmaxf(v0, v1);
            #pragma unroll
            for (int msk = 1; msk < 16; msk <<= 1)
                mt = fmaxf(mt, __shfl_xor(mt, msk, 32));
            const float mnew = fmaxf(mrow[r], mt);
            const float c = __expf(mrow[r] - mnew);
            mrow[r] = mnew;
            const float e0 = __expf(v0 - mnew);
            const float e1 = __expf(v1 - mnew);
            float ps = e0 + e1;
            #pragma unroll
            for (int msk = 1; msk < 16; msk <<= 1)
                ps += __shfl_xor(ps, msk, 32);
            lrow[r] = lrow[r] * c + ps;
            csc[r] = c; p0[r] = e0; p1[r] = e1;
        }

        // stage P (C-layout) -> LDS -> reload as A-operand (contiguous chunks)
        #pragma unroll
        for (int r = 0; r < 8; ++r) {
            const int qm = r + mOff;
            pstage[wib][qm * 32 + n16]      = (__bf16)p0[r];
            pstage[wib][qm * 32 + 16 + n16] = (__bf16)p1[r];
        }
        __builtin_amdgcn_wave_barrier();
        asm volatile("s_wait_dscnt 0" ::: "memory");
        __builtin_amdgcn_wave_barrier();
        const __bf16* pp = &pstage[wib][n16 * 32 + aKh];
        const v16bf pa = cat16(*(const v8bf*)pp, *(const v8bf*)(pp + 8));
        __builtin_amdgcn_wave_barrier();

        // rescale accumulators, then acc += P @ V  (V^T makes B-operand contiguous)
        #pragma unroll
        for (int c4 = 0; c4 < 4; ++c4) {
            #pragma unroll
            for (int r = 0; r < 8; ++r) acc[c4][r] *= csc[r];
        }
        #pragma unroll
        for (int c4 = 0; c4 < 4; ++c4) {
            const int n = c4 * 16 + n16;
            const v16bf vb = *(const v16bf*)(Vp + (size_t)n * Sn + k0 + bKh);
            acc[c4] = wmma_bf16(pa, vb, acc[c4]);
        }
    }

    // normalize + gathered store (valid rows of batch b contiguous: row = cu[b]+s)
    #pragma unroll
    for (int r = 0; r < 8; ++r) {
        const int s = q0 + r + mOff;
        if (s >= len) continue;
        const float inv = 1.0f / lrow[r];
        const size_t orow = (size_t)(base + s) * HIDn + h * Dn;
        #pragma unroll
        for (int c4 = 0; c4 < 4; ++c4)
            out[orow + c4 * 16 + n16] = acc[c4][r] * inv;
    }
}

extern "C" void kernel_launch(void* const* d_in, const int* in_sizes, int n_in,
                              void* d_out, int out_size, void* d_ws, size_t ws_size,
                              hipStream_t stream) {
    const float* X   = (const float*)d_in[0];
    const float* W   = (const float*)d_in[1];
    const float* bW  = (const float*)d_in[2];
    const float* bia = (const float*)d_in[3];
    const int*   idx = (const int*)d_in[4];
    const int*   cu  = (const int*)d_in[6];
    float*       out = (float*)d_out;

    const size_t per = (size_t)Bn * Hn * Sn * Dn;      // 6,291,456 elems
    char* wsc = (char*)d_ws;                            // total ~51.3 MB
    __bf16* Qb = (__bf16*)(wsc);
    __bf16* Kb = (__bf16*)(wsc + per * 2);
    __bf16* Vt = (__bf16*)(wsc + per * 4);
    __bf16* Xb = (__bf16*)(wsc + per * 6);                          // 10,027,008 B
    __bf16* Wt = (__bf16*)(wsc + per * 6 + (size_t)NNZ * HIDn * 2); //  3,538,944 B

    hipMemsetAsync(d_ws, 0, per * 3 * sizeof(__bf16), stream);

    dim3 tg(NCOL / 32, HIDn / 32);
    prep_w<<<tg, 256, 0, stream>>>(W, Wt);
    prep_x<<<(NNZ * HIDn / 4) / 256, 256, 0, stream>>>(X, Xb);
    qkv_gemm<<<1836, 256, 0, stream>>>(Xb, Wt, bW, idx, Qb, Kb, Vt);
    attn<<<768, 256, 0, stream>>>(Qb, Kb, Vt, bia, cu, out);
}
